// SparseMoe_21363167330578
// MI455X (gfx1250) — matrix operand order repaired and checked by reading, not compiled, not run on hardware
//
#include <hip/hip_runtime.h>
#include <cstdint>
#include <cstddef>

// Problem constants (match reference setup_inputs)
#define T_TOK 8192
#define D_DIM 1024
#define E_EXP 16
#define I_DIM 512
#define SI_DIM 2048
#define MAXROWS (2 * T_TOK + E_EXP * 32)
#define KSTAGE 512          // K elems staged in LDS per round
#define APITCH (KSTAGE + 8) // LDS row pitch (bf16): row stride = 260 dwords -> +4 banks/row

typedef __attribute__((ext_vector_type(16))) __bf16 v16bf;
typedef __attribute__((ext_vector_type(8)))  __bf16 v8bf;
typedef __attribute__((ext_vector_type(8)))  float  v8f;

__device__ __forceinline__ v8f wmma_bf16(v16bf a, v16bf b, v8f c) {
  // (neg_a, A, neg_b, B, c_mod, C, reuse_a, reuse_b)
  return __builtin_amdgcn_wmma_f32_16x16x32_bf16(false, a, false, b, (short)0, c,
                                                 false, false);
}

// A-operand (16-bit A 16x32): lane holds K = kh*8..+7 and +16..+23 of its row
__device__ __forceinline__ v16bf load_a16(const __bf16* p) {
  v8bf lo = *reinterpret_cast<const v8bf*>(p);
  v8bf hi = *reinterpret_cast<const v8bf*>(p + 16);
  return __builtin_shufflevector(lo, hi, 0, 1, 2, 3, 4, 5, 6, 7, 8, 9, 10, 11, 12, 13, 14, 15);
}
// B-operand (16-bit B 32x16): lane holds K = kh*8..+7 and +16..+23 of column lane%16
__device__ __forceinline__ v16bf load_b16(const __bf16* __restrict__ p) {
  v8bf lo = *reinterpret_cast<const v8bf*>(p);
  v8bf hi = *reinterpret_cast<const v8bf*>(p + 16);
  return __builtin_shufflevector(lo, hi, 0, 1, 2, 3, 4, 5, 6, 7, 8, 9, 10, 11, 12, 13, 14, 15);
}

__device__ __forceinline__ float silu(float g) { return g / (1.0f + __expf(-g)); }

// ---------------- conversion kernels ----------------

__global__ __launch_bounds__(256) void cvt_bf16_kernel(const float* __restrict__ in,
                                                       __bf16* __restrict__ out, int n) {
  int i = blockIdx.x * 256 + threadIdx.x;
  if (i < n) out[i] = (__bf16)in[i];
}

// in: fp32 [R, C] (batched, stride R*C), out: bf16 [C, R] (batched, stride R*C)
__global__ __launch_bounds__(256) void cvt_trans_kernel(const float* __restrict__ in,
                                                        __bf16* __restrict__ out,
                                                        int R, int C) {
  __shared__ float tile[32][33];
  size_t base = (size_t)blockIdx.z * R * C;
  int x0 = blockIdx.x * 32, y0 = blockIdx.y * 32;
  int tx = threadIdx.x & 31, ty = threadIdx.x >> 5;  // 32 x 8
#pragma unroll
  for (int i = 0; i < 4; ++i) {
    int row = y0 + ty + i * 8;
    tile[ty + i * 8][tx] = in[base + (size_t)row * C + x0 + tx];
  }
  __syncthreads();
#pragma unroll
  for (int i = 0; i < 4; ++i) {
    int orow = x0 + ty + i * 8;
    out[base + (size_t)orow * R + y0 + tx] = (__bf16)tile[tx][ty + i * 8];
  }
}

// ---------------- router ----------------

__global__ __launch_bounds__(256) void router_kernel(const float* __restrict__ x,
                                                     const float* __restrict__ gate_w,
                                                     const float* __restrict__ segw,
                                                     float* __restrict__ logits_out,
                                                     int* __restrict__ tidx,
                                                     float* __restrict__ tw,
                                                     float* __restrict__ sig) {
  __shared__ float slog[8][16];
  int wave = threadIdx.x >> 5, lane = threadIdx.x & 31;
  int t = blockIdx.x * 8 + wave;
  float acc[E_EXP];
#pragma unroll
  for (int e = 0; e < E_EXP; ++e) acc[e] = 0.0f;
  float sacc = 0.0f;
  for (int i = lane; i < D_DIM; i += 32) {
    float xv = x[(size_t)t * D_DIM + i];
    sacc += xv * segw[i];
    const float* gr = gate_w + (size_t)i * E_EXP;
#pragma unroll
    for (int e = 0; e < E_EXP; ++e) acc[e] += xv * gr[e];
  }
#pragma unroll
  for (int e = 0; e < E_EXP; ++e)
    for (int o = 16; o >= 1; o >>= 1) acc[e] += __shfl_xor(acc[e], o, 32);
  for (int o = 16; o >= 1; o >>= 1) sacc += __shfl_xor(sacc, o, 32);
#pragma unroll
  for (int e = 0; e < E_EXP; ++e) {
    if (lane == e) {
      logits_out[(size_t)t * E_EXP + e] = acc[e];
      slog[wave][e] = acc[e];
    }
  }
  __syncthreads();
  if (lane == 0) {
    int i1 = 0;
    float l1 = slog[wave][0];
    for (int e = 1; e < E_EXP; ++e)
      if (slog[wave][e] > l1) { l1 = slog[wave][e]; i1 = e; }
    int i2 = (i1 == 0) ? 1 : 0;
    float l2 = slog[wave][i2];
    for (int e = 0; e < E_EXP; ++e)
      if (e != i1 && slog[wave][e] > l2) { l2 = slog[wave][e]; i2 = e; }
    // normalized top-2 softmax weights == sigmoid of logit difference
    float w1 = 1.0f / (1.0f + __expf(l2 - l1));
    tidx[2 * t] = i1;
    tidx[2 * t + 1] = i2;
    tw[2 * t] = w1;
    tw[2 * t + 1] = 1.0f - w1;
    sig[t] = 1.0f / (1.0f + __expf(-sacc));
  }
}

// ---------------- routing compaction ----------------

__global__ __launch_bounds__(256) void count_kernel(const int* __restrict__ tidx,
                                                    int* __restrict__ counts) {
  int t = blockIdx.x * 256 + threadIdx.x;
  if (t >= T_TOK) return;
  atomicAdd(&counts[tidx[2 * t]], 1);
  atomicAdd(&counts[tidx[2 * t + 1]], 1);
}

__global__ void scan_kernel(const int* __restrict__ counts, int* __restrict__ off) {
  if (threadIdx.x == 0 && blockIdx.x == 0) {
    int acc = 0;
    for (int e = 0; e < E_EXP; ++e) {
      off[e] = acc;
      acc += (counts[e] + 31) & ~31;  // pad each segment to a 32-row M-tile
    }
    off[E_EXP] = acc;
  }
}

__global__ __launch_bounds__(256) void scatter_kernel(const int* __restrict__ tidx,
                                                      const float* __restrict__ tw,
                                                      const int* __restrict__ off,
                                                      int* __restrict__ counts2,
                                                      int* __restrict__ row_tok,
                                                      float* __restrict__ row_w) {
  int t = blockIdx.x * 256 + threadIdx.x;
  if (t >= T_TOK) return;
#pragma unroll
  for (int k = 0; k < 2; ++k) {
    int e = tidx[2 * t + k];
    int j = atomicAdd(&counts2[e], 1);
    int r = off[e] + j;
    row_tok[r] = t;
    row_w[r] = tw[2 * t + k];
  }
}

__global__ void pad_kernel(const int* __restrict__ counts, const int* __restrict__ off,
                           int* __restrict__ row_tok, float* __restrict__ row_w) {
  int e = blockIdx.x;
  int cnt = counts[e];
  int padded = (cnt + 31) & ~31;
  for (int j = cnt + threadIdx.x; j < padded; j += 32) {
    row_tok[off[e] + j] = 0;
    row_w[off[e] + j] = 0.0f;
  }
}

// ---------------- expert gate/up: LDS-staged A, 32M x 32N/wave, gate+up fused ----------
// grid: (I/256, T/32, E), block 256 (8 waves)
__global__ __launch_bounds__(256) void moe_gateup_kernel(
    const __bf16* __restrict__ xb, const __bf16* __restrict__ wgT,
    const __bf16* __restrict__ wuT, const int* __restrict__ off,
    const int* __restrict__ row_tok, __bf16* __restrict__ act) {
  __shared__ __attribute__((aligned(16))) __bf16 As[32 * APITCH];
  const int e = blockIdx.z;
  const int base = off[e];
  if ((int)blockIdx.y * 32 >= off[e + 1] - base) return;
  const int rowbase = base + blockIdx.y * 32;
  const int lane = threadIdx.x & 31;
  const int wave = threadIdx.x >> 5;
  const int n0 = blockIdx.x * 256 + wave * 32;
  const int m = lane & 15;
  const int kh = lane >> 4;
  const size_t wb = (size_t)e * I_DIM * D_DIM;
  const __bf16* bg0 = wgT + wb + (size_t)(n0 + m) * D_DIM;
  const __bf16* bg1 = bg0 + (size_t)16 * D_DIM;
  const __bf16* bu0 = wuT + wb + (size_t)(n0 + m) * D_DIM;
  const __bf16* bu1 = bu0 + (size_t)16 * D_DIM;
  const __bf16* a0p = &As[m * APITCH];
  const __bf16* a1p = &As[(m + 16) * APITCH];
  v8f cg00 = {}, cg01 = {}, cu00 = {}, cu01 = {};
  v8f cg10 = {}, cg11 = {}, cu10 = {}, cu11 = {};
  for (int st = 0; st < D_DIM / KSTAGE; ++st) {
    if (st) __syncthreads();
    const int k0 = st * KSTAGE;
    for (int c = threadIdx.x; c < 32 * (KSTAGE / 8); c += 256) {
      int row = c >> 6;             // KSTAGE/8 == 64 chunks per row
      int kb = (c & 63) * 8;
      int tok = row_tok[rowbase + row];
      *reinterpret_cast<v8bf*>(&As[row * APITCH + kb]) =
          *reinterpret_cast<const v8bf*>(xb + (size_t)tok * D_DIM + k0 + kb);
    }
    __syncthreads();
    for (int kc = 0; kc < KSTAGE / 32; ++kc) {
      v16bf a0 = load_a16(a0p + kc * 32 + kh * 8);
      v16bf a1 = load_a16(a1p + kc * 32 + kh * 8);
      int kb2 = k0 + kc * 32 + kh * 8;
      if (kb2 + 256 < D_DIM) __builtin_prefetch(bg0 + kb2 + 256, 0, 3);
      v16bf b = load_b16(bg0 + kb2);
      cg00 = wmma_bf16(a0, b, cg00);
      cg10 = wmma_bf16(a1, b, cg10);
      b = load_b16(bg1 + kb2);
      cg01 = wmma_bf16(a0, b, cg01);
      cg11 = wmma_bf16(a1, b, cg11);
      b = load_b16(bu0 + kb2);
      cu00 = wmma_bf16(a0, b, cu00);
      cu10 = wmma_bf16(a1, b, cu10);
      b = load_b16(bu1 + kb2);
      cu01 = wmma_bf16(a0, b, cu01);
      cu11 = wmma_bf16(a1, b, cu11);
    }
  }
  const int col = n0 + (lane & 15);
#pragma unroll
  for (int i = 0; i < 8; ++i) {
    int mm = i + 8 * kh;  // C layout: M = i + 8*(lane/16)
    size_t r0 = (size_t)(rowbase + mm) * I_DIM;
    size_t r1 = (size_t)(rowbase + 16 + mm) * I_DIM;
    act[r0 + col] = (__bf16)(silu(cg00[i]) * cu00[i]);
    act[r0 + col + 16] = (__bf16)(silu(cg01[i]) * cu01[i]);
    act[r1 + col] = (__bf16)(silu(cg10[i]) * cu10[i]);
    act[r1 + col + 16] = (__bf16)(silu(cg11[i]) * cu11[i]);
  }
}

// ---------------- expert down: LDS-staged A, 32M x 64N/wave, atomic scatter-add --------
// grid: (D/512, MAXROWS/32), block 256
__global__ __launch_bounds__(256) void moe_down_kernel(
    const __bf16* __restrict__ act, const __bf16* __restrict__ wdT,
    const int* __restrict__ off, const int* __restrict__ row_tok,
    const float* __restrict__ row_w, float* __restrict__ out) {
  __shared__ __attribute__((aligned(16))) __bf16 As[32 * APITCH];
  const int r0 = blockIdx.y * 32;
  if (r0 >= off[E_EXP]) return;
  int e = 0;
#pragma unroll
  for (int i = 0; i < E_EXP; ++i)
    if (r0 >= off[i] && r0 < off[i + 1]) e = i;  // 32-aligned tiles never straddle experts
  for (int c = threadIdx.x; c < 32 * (I_DIM / 8); c += 256) {
    int row = c >> 6;
    int kb = (c & 63) * 8;
    *reinterpret_cast<v8bf*>(&As[row * APITCH + kb]) =
        *reinterpret_cast<const v8bf*>(act + (size_t)(r0 + row) * I_DIM + kb);
  }
  __syncthreads();
  const int lane = threadIdx.x & 31;
  const int wave = threadIdx.x >> 5;
  const int n0 = blockIdx.x * 512 + wave * 64;
  const int m = lane & 15;
  const int kh = lane >> 4;
  const size_t wb = (size_t)e * D_DIM * I_DIM;
  const __bf16* b0 = wdT + wb + (size_t)(n0 + m) * I_DIM;
  const __bf16* b1 = b0 + (size_t)16 * I_DIM;
  const __bf16* b2 = b0 + (size_t)32 * I_DIM;
  const __bf16* b3 = b0 + (size_t)48 * I_DIM;
  const __bf16* a0p = &As[m * APITCH];
  const __bf16* a1p = &As[(m + 16) * APITCH];
  v8f c00 = {}, c01 = {}, c02 = {}, c03 = {};
  v8f c10 = {}, c11 = {}, c12 = {}, c13 = {};
  for (int kc = 0; kc < I_DIM / 32; ++kc) {
    v16bf a0 = load_a16(a0p + kc * 32 + kh * 8);
    v16bf a1 = load_a16(a1p + kc * 32 + kh * 8);
    int kb2 = kc * 32 + kh * 8;
    if (kb2 + 256 < I_DIM) __builtin_prefetch(b0 + kb2 + 256, 0, 3);
    v16bf b = load_b16(b0 + kb2);
    c00 = wmma_bf16(a0, b, c00);
    c10 = wmma_bf16(a1, b, c10);
    b = load_b16(b1 + kb2);
    c01 = wmma_bf16(a0, b, c01);
    c11 = wmma_bf16(a1, b, c11);
    b = load_b16(b2 + kb2);
    c02 = wmma_bf16(a0, b, c02);
    c12 = wmma_bf16(a1, b, c12);
    b = load_b16(b3 + kb2);
    c03 = wmma_bf16(a0, b, c03);
    c13 = wmma_bf16(a1, b, c13);
  }
  const int col = n0 + (lane & 15);
#pragma unroll
  for (int i = 0; i < 8; ++i) {
    int ra = r0 + i + 8 * kh;
    int rb = ra + 16;
    int ta = row_tok[ra];
    int tb = row_tok[rb];
    float wa = row_w[ra];
    float wb2 = row_w[rb];
    float* oa = out + (size_t)ta * D_DIM + col;
    float* ob = out + (size_t)tb * D_DIM + col;
    atomicAdd(oa + 0, wa * c00[i]);
    atomicAdd(oa + 16, wa * c01[i]);
    atomicAdd(oa + 32, wa * c02[i]);
    atomicAdd(oa + 48, wa * c03[i]);
    atomicAdd(ob + 0, wb2 * c10[i]);
    atomicAdd(ob + 16, wb2 * c11[i]);
    atomicAdd(ob + 32, wb2 * c12[i]);
    atomicAdd(ob + 48, wb2 * c13[i]);
  }
}

// ---------------- shared expert gate/up: LDS-staged A, 32M ----------------
// grid: (SI/256, T/32), block 256
__global__ __launch_bounds__(256) void shared_gateup_kernel(
    const __bf16* __restrict__ xb, const __bf16* __restrict__ wgT,
    const __bf16* __restrict__ wuT, __bf16* __restrict__ act) {
  __shared__ __attribute__((aligned(16))) __bf16 As[32 * APITCH];
  const int rowbase = blockIdx.y * 32;
  const int lane = threadIdx.x & 31;
  const int wave = threadIdx.x >> 5;
  const int n0 = blockIdx.x * 256 + wave * 32;
  const int m = lane & 15;
  const int kh = lane >> 4;
  const __bf16* bg0 = wgT + (size_t)(n0 + m) * D_DIM;
  const __bf16* bg1 = bg0 + (size_t)16 * D_DIM;
  const __bf16* bu0 = wuT + (size_t)(n0 + m) * D_DIM;
  const __bf16* bu1 = bu0 + (size_t)16 * D_DIM;
  const __bf16* a0p = &As[m * APITCH];
  const __bf16* a1p = &As[(m + 16) * APITCH];
  v8f cg00 = {}, cg01 = {}, cu00 = {}, cu01 = {};
  v8f cg10 = {}, cg11 = {}, cu10 = {}, cu11 = {};
  for (int st = 0; st < D_DIM / KSTAGE; ++st) {
    if (st) __syncthreads();
    const int k0 = st * KSTAGE;
    for (int c = threadIdx.x; c < 32 * (KSTAGE / 8); c += 256) {
      int row = c >> 6;
      int kb = (c & 63) * 8;
      *reinterpret_cast<v8bf*>(&As[row * APITCH + kb]) =
          *reinterpret_cast<const v8bf*>(xb + (size_t)(rowbase + row) * D_DIM + k0 + kb);
    }
    __syncthreads();
    for (int kc = 0; kc < KSTAGE / 32; ++kc) {
      v16bf a0 = load_a16(a0p + kc * 32 + kh * 8);
      v16bf a1 = load_a16(a1p + kc * 32 + kh * 8);
      int kb2 = k0 + kc * 32 + kh * 8;
      if (kb2 + 256 < D_DIM) __builtin_prefetch(bg0 + kb2 + 256, 0, 3);
      v16bf b = load_b16(bg0 + kb2);
      cg00 = wmma_bf16(a0, b, cg00);
      cg10 = wmma_bf16(a1, b, cg10);
      b = load_b16(bg1 + kb2);
      cg01 = wmma_bf16(a0, b, cg01);
      cg11 = wmma_bf16(a1, b, cg11);
      b = load_b16(bu0 + kb2);
      cu00 = wmma_bf16(a0, b, cu00);
      cu10 = wmma_bf16(a1, b, cu10);
      b = load_b16(bu1 + kb2);
      cu01 = wmma_bf16(a0, b, cu01);
      cu11 = wmma_bf16(a1, b, cu11);
    }
  }
  const int col = n0 + (lane & 15);
#pragma unroll
  for (int i = 0; i < 8; ++i) {
    int mm = i + 8 * kh;
    size_t r0 = (size_t)(rowbase + mm) * SI_DIM;
    size_t r1 = (size_t)(rowbase + 16 + mm) * SI_DIM;
    act[r0 + col] = (__bf16)(silu(cg00[i]) * cu00[i]);
    act[r0 + col + 16] = (__bf16)(silu(cg01[i]) * cu01[i]);
    act[r1 + col] = (__bf16)(silu(cg10[i]) * cu10[i]);
    act[r1 + col + 16] = (__bf16)(silu(cg11[i]) * cu11[i]);
  }
}

// ---------------- shared expert down: 4-stage LDS A (K=2048), initializes out ----------
// grid: (D/512, T/32), block 256
__global__ __launch_bounds__(256) void shared_down_kernel(
    const __bf16* __restrict__ act, const __bf16* __restrict__ wdT,
    const float* __restrict__ sig, float* __restrict__ out) {
  __shared__ __attribute__((aligned(16))) __bf16 As[32 * APITCH];
  const int rowbase = blockIdx.y * 32;
  const int lane = threadIdx.x & 31;
  const int wave = threadIdx.x >> 5;
  const int n0 = blockIdx.x * 512 + wave * 64;
  const int m = lane & 15;
  const int kh = lane >> 4;
  const __bf16* b0 = wdT + (size_t)(n0 + m) * SI_DIM;
  const __bf16* b1 = b0 + (size_t)16 * SI_DIM;
  const __bf16* b2 = b0 + (size_t)32 * SI_DIM;
  const __bf16* b3 = b0 + (size_t)48 * SI_DIM;
  const __bf16* a0p = &As[m * APITCH];
  const __bf16* a1p = &As[(m + 16) * APITCH];
  v8f c00 = {}, c01 = {}, c02 = {}, c03 = {};
  v8f c10 = {}, c11 = {}, c12 = {}, c13 = {};
  for (int st = 0; st < SI_DIM / KSTAGE; ++st) {
    if (st) __syncthreads();
    const int k0 = st * KSTAGE;
    for (int c = threadIdx.x; c < 32 * (KSTAGE / 8); c += 256) {
      int row = c >> 6;
      int kb = (c & 63) * 8;
      *reinterpret_cast<v8bf*>(&As[row * APITCH + kb]) =
          *reinterpret_cast<const v8bf*>(act + (size_t)(rowbase + row) * SI_DIM + k0 + kb);
    }
    __syncthreads();
    for (int kc = 0; kc < KSTAGE / 32; ++kc) {
      v16bf a0 = load_a16(a0p + kc * 32 + kh * 8);
      v16bf a1 = load_a16(a1p + kc * 32 + kh * 8);
      int kb2 = k0 + kc * 32 + kh * 8;
      if (kb2 + 256 < SI_DIM) __builtin_prefetch(b0 + kb2 + 256, 0, 3);
      v16bf b = load_b16(b0 + kb2);
      c00 = wmma_bf16(a0, b, c00);
      c10 = wmma_bf16(a1, b, c10);
      b = load_b16(b1 + kb2);
      c01 = wmma_bf16(a0, b, c01);
      c11 = wmma_bf16(a1, b, c11);
      b = load_b16(b2 + kb2);
      c02 = wmma_bf16(a0, b, c02);
      c12 = wmma_bf16(a1, b, c12);
      b = load_b16(b3 + kb2);
      c03 = wmma_bf16(a0, b, c03);
      c13 = wmma_bf16(a1, b, c13);
    }
  }
  const int col = n0 + (lane & 15);
#pragma unroll
  for (int i = 0; i < 8; ++i) {
    int ta = rowbase + i + 8 * kh;
    int tb = ta + 16;
    float sa = sig[ta];
    float sb = sig[tb];
    float* oa = out + (size_t)ta * D_DIM + col;
    float* ob = out + (size_t)tb * D_DIM + col;
    oa[0] = sa * c00[i];                 // full coverage: initializes out
    oa[16] = sa * c01[i];
    oa[32] = sa * c02[i];
    oa[48] = sa * c03[i];
    ob[0] = sb * c10[i];
    ob[16] = sb * c11[i];
    ob[32] = sb * c12[i];
    ob[48] = sb * c13[i];
  }
}

// ---------------- host side ----------------

extern "C" void kernel_launch(void* const* d_in, const int* in_sizes, int n_in,
                              void* d_out, int out_size, void* d_ws, size_t ws_size,
                              hipStream_t stream) {
  (void)in_sizes; (void)n_in; (void)out_size; (void)ws_size;
  const float* x    = (const float*)d_in[0];
  const float* gw   = (const float*)d_in[1];
  const float* egw  = (const float*)d_in[2];
  const float* euw  = (const float*)d_in[3];
  const float* edw  = (const float*)d_in[4];
  const float* sgw  = (const float*)d_in[5];
  const float* suw  = (const float*)d_in[6];
  const float* sdw  = (const float*)d_in[7];
  const float* segw = (const float*)d_in[8];

  float* out = (float*)d_out;
  float* logits_out = out + (size_t)T_TOK * D_DIM;

  uint8_t* p = (uint8_t*)d_ws;
  auto alloc = [&](size_t bytes) -> void* {
    void* r = (void*)p;
    p += (bytes + 255) & ~(size_t)255;
    return r;
  };
  __bf16* xb   = (__bf16*)alloc((size_t)T_TOK * D_DIM * 2);
  __bf16* egT  = (__bf16*)alloc((size_t)E_EXP * I_DIM * D_DIM * 2);   // [e][I][D]
  __bf16* euT  = (__bf16*)alloc((size_t)E_EXP * I_DIM * D_DIM * 2);   // [e][I][D]
  __bf16* edT  = (__bf16*)alloc((size_t)E_EXP * D_DIM * I_DIM * 2);   // [e][D][I]
  __bf16* sgT  = (__bf16*)alloc((size_t)SI_DIM * D_DIM * 2);          // [SI][D]
  __bf16* suT  = (__bf16*)alloc((size_t)SI_DIM * D_DIM * 2);          // [SI][D]
  __bf16* sdT  = (__bf16*)alloc((size_t)D_DIM * SI_DIM * 2);          // [D][SI]
  __bf16* actS = (__bf16*)alloc((size_t)T_TOK * SI_DIM * 2);
  __bf16* actE = (__bf16*)alloc((size_t)MAXROWS * I_DIM * 2);
  float*  sig  = (float*)alloc((size_t)T_TOK * 4);
  int*    tidx = (int*)alloc((size_t)T_TOK * 2 * 4);
  float*  tw   = (float*)alloc((size_t)T_TOK * 2 * 4);
  int*    counts  = (int*)alloc(256);
  int*    counts2 = (int*)alloc(256);
  int*    off     = (int*)alloc(256);
  int*    row_tok = (int*)alloc((size_t)MAXROWS * 4);
  float*  row_w   = (float*)alloc((size_t)MAXROWS * 4);

  hipMemsetAsync(counts, 0, 256, stream);
  hipMemsetAsync(counts2, 0, 256, stream);

  // 1) precision conversion + weight transposition (N-major for WMMA B-operand)
  cvt_bf16_kernel<<<(T_TOK * D_DIM) / 256, 256, 0, stream>>>(x, xb, T_TOK * D_DIM);
  cvt_trans_kernel<<<dim3(I_DIM / 32, D_DIM / 32, E_EXP), 256, 0, stream>>>(egw, egT, D_DIM, I_DIM);
  cvt_trans_kernel<<<dim3(I_DIM / 32, D_DIM / 32, E_EXP), 256, 0, stream>>>(euw, euT, D_DIM, I_DIM);
  cvt_trans_kernel<<<dim3(D_DIM / 32, I_DIM / 32, E_EXP), 256, 0, stream>>>(edw, edT, I_DIM, D_DIM);
  cvt_trans_kernel<<<dim3(SI_DIM / 32, D_DIM / 32, 1), 256, 0, stream>>>(sgw, sgT, D_DIM, SI_DIM);
  cvt_trans_kernel<<<dim3(SI_DIM / 32, D_DIM / 32, 1), 256, 0, stream>>>(suw, suT, D_DIM, SI_DIM);
  cvt_trans_kernel<<<dim3(D_DIM / 32, SI_DIM / 32, 1), 256, 0, stream>>>(sdw, sdT, SI_DIM, D_DIM);

  // 2) routing
  router_kernel<<<T_TOK / 8, 256, 0, stream>>>(x, gw, segw, logits_out, tidx, tw, sig);
  count_kernel<<<T_TOK / 256, 256, 0, stream>>>(tidx, counts);
  scan_kernel<<<1, 32, 0, stream>>>(counts, off);
  scatter_kernel<<<T_TOK / 256, 256, 0, stream>>>(tidx, tw, off, counts2, row_tok, row_w);
  pad_kernel<<<E_EXP, 32, 0, stream>>>(counts, off, row_tok, row_w);

  // 3) shared expert (down pass initializes out)
  shared_gateup_kernel<<<dim3(SI_DIM / 256, T_TOK / 32, 1), 256, 0, stream>>>(xb, sgT, suT, actS);
  shared_down_kernel<<<dim3(D_DIM / 512, T_TOK / 32, 1), 256, 0, stream>>>(actS, sdT, sig, out);

  // 4) routed experts (down pass atomic-adds into out)
  const int maxTiles = T_TOK / 32;                 // per-expert worst case
  const int maxPaddedTiles = MAXROWS / 32;         // global worst case
  moe_gateup_kernel<<<dim3(I_DIM / 256, maxTiles, E_EXP), 256, 0, stream>>>(
      xb, egT, euT, off, row_tok, actE);
  moe_down_kernel<<<dim3(D_DIM / 512, maxPaddedTiles, 1), 256, 0, stream>>>(
      actE, edT, off, row_tok, row_w, out);
}
